// LSTMNet_10428180594912
// MI455X (gfx1250) — matrix operand order repaired and checked by reading, not compiled, not run on hardware
//
#include <hip/hip_runtime.h>
#include <hip/hip_bf16.h>

typedef __attribute__((ext_vector_type(16))) __bf16 v16bf;
typedef __attribute__((ext_vector_type(8)))  float  v8f;
typedef __attribute__((ext_vector_type(8)))  int    v8i;

#define AROW 1032  // LDS h-tile row stride in ushorts (516 dwords, ==4 mod 64 -> balanced banks)

union Op32 {        // 32-byte WMMA operand assembled from two 16B loads
  int4  q[2];
  v8i   i;
  v16bf b;
};

__device__ __forceinline__ unsigned short f2bf(float f) {  // f32 -> bf16 (RNE)
  unsigned u = __float_as_uint(f);
  return (unsigned short)((u + 0x7FFFu + ((u >> 16) & 1u)) >> 16);
}
__device__ __forceinline__ float bf2f(unsigned short h) {
  return __uint_as_float(((unsigned)h) << 16);
}
__device__ __forceinline__ float sigm(float v) {
  return 1.0f / (1.0f + __expf(-v));
}
__device__ __forceinline__ float tanh_fast(float v) {  // overflow-safe tanh
  float s = 1.0f / (1.0f + __expf(-2.0f * v));
  return 2.0f * s - 1.0f;
}

// ---------------------------------------------------------------------------
// Pack W_hh [4096,1024] into bf16 hi/lo pair in WMMA B-operand (32x16) order:
//   Wp[tn=256][kk=32][lane=32][16 bf16], lane: n=lane&15 (g=tn*16+n),
//   K-run = kk*32 + 16*(lane>>4) + 0..15 -> two contiguous b128 per lane.
// ---------------------------------------------------------------------------
__global__ void lstm_pack(const float* __restrict__ Whh,
                          unsigned short* __restrict__ Wphi,
                          unsigned short* __restrict__ Wplo) {
  int e = blockIdx.x * blockDim.x + threadIdx.x;  // 0 .. 262143
  int tn   = e >> 10;
  int rem  = e & 1023;
  int kk   = rem >> 5;
  int lane = rem & 31;
  int g  = tn * 16 + (lane & 15);
  int kb = kk * 32 + 16 * (lane >> 4);
  const float* w = Whh + g * 1024 + kb;
  unsigned hi[8], lo[8];
#pragma unroll
  for (int i = 0; i < 8; ++i) {
    float f0 = w[2 * i], f1 = w[2 * i + 1];
    unsigned short h0 = f2bf(f0), h1 = f2bf(f1);
    unsigned short l0 = f2bf(f0 - bf2f(h0)), l1 = f2bf(f1 - bf2f(h1));
    hi[i] = (unsigned)h0 | ((unsigned)h1 << 16);
    lo[i] = (unsigned)l0 | ((unsigned)l1 << 16);
  }
  int4* dh = (int4*)(Wphi + (size_t)e * 16);
  int4* dl = (int4*)(Wplo + (size_t)e * 16);
  dh[0] = make_int4(hi[0], hi[1], hi[2], hi[3]);
  dh[1] = make_int4(hi[4], hi[5], hi[6], hi[7]);
  dl[0] = make_int4(lo[0], lo[1], lo[2], lo[3]);
  dl[1] = make_int4(lo[4], lo[5], lo[6], lo[7]);
}

// bsum = b_ih + b_hh; split h0 into bf16 hi/lo; copy c0
__global__ void lstm_init(const float* __restrict__ bih, const float* __restrict__ bhh,
                          const float* __restrict__ h0, const float* __restrict__ c0,
                          float* __restrict__ bsum,
                          unsigned short* __restrict__ hhiA,
                          unsigned short* __restrict__ hloA,
                          float* __restrict__ cb) {
  int i = blockIdx.x * blockDim.x + threadIdx.x;
  if (i < 4096) bsum[i] = bih[i] + bhh[i];
  int j = i - 4096;
  if (j >= 0 && j < 65536) {
    float v = h0[j];
    unsigned short hh = f2bf(v);
    hhiA[j] = hh;
    hloA[j] = f2bf(v - bf2f(hh));
  }
  int k = i - 4096 - 65536;
  if (k >= 0 && k < 65536) cb[k] = c0[k];
}

// ---------------------------------------------------------------------------
// One LSTM timestep; gates = hhi@Whi + hhi@Wlo + hlo@Whi (3xBF16 ~= FP32).
// Blocks 0..255: (btile, jt) tile; 4 waves = gates i,f,g,o, each 32 K-slices
//   x 3 V_WMMA_F32_16X16X32_BF16. Blocks 256..259: head out[t-1].
// ---------------------------------------------------------------------------
__global__ __launch_bounds__(128) void lstm_step(
    const float* __restrict__ x,            // [512][64]
    const unsigned short* __restrict__ Wphi,
    const unsigned short* __restrict__ Wplo,
    const float* __restrict__ bsum,         // [4096]
    const float* __restrict__ Wih,          // [4096]
    float* __restrict__ cbuf,               // [64][1024] (in-place)
    const unsigned short* __restrict__ hhiP,
    const unsigned short* __restrict__ hloP,
    unsigned short* __restrict__ hhiN,
    unsigned short* __restrict__ hloN,
    const float* __restrict__ Wout,         // [1024]
    const float* __restrict__ bout,         // [1]
    float* __restrict__ out,                // [512][64]
    int t) {
  __shared__ unsigned short sAhi[16 * AROW];
  __shared__ unsigned short sAlo[16 * AROW];
  __shared__ float sG[4 * 256];
  const int tid = threadIdx.x;

  if (blockIdx.x >= 256) {
    // ---- output head for the PREVIOUS step's h (hhiP/hloP are h_{t-1}) ----
    if (t == 0) return;
    int bt = blockIdx.x - 256;
    int bl = tid >> 3, sub = tid & 7;
    int b = bt * 16 + bl;
    const unsigned short* hh = hhiP + b * 1024;
    const unsigned short* hl = hloP + b * 1024;
    float s = 0.f;
    for (int j = sub; j < 1024; j += 8)
      s += (bf2f(hh[j]) + bf2f(hl[j])) * Wout[j];
    sG[tid] = s;
    __syncthreads();
    if (sub == 0) {
      float tot = 0.f;
#pragma unroll
      for (int q = 0; q < 8; ++q) tot += sG[bl * 8 + q];
      out[(t - 1) * 64 + b] = tot + bout[0];
    }
    return;
  }

  const int wg = blockIdx.x;
  const int btile = wg >> 6;
  const int jt = wg & 63;
  const int wave = tid >> 5;
  const int lane = tid & 31;

  // ---- stage hhi/hlo tiles [16][1024] bf16 into padded LDS (b128 copies) ----
  {
    const int4* ghi = (const int4*)hhiP + (size_t)btile * 2048;  // 16 rows * 128 int4
    const int4* glo = (const int4*)hloP + (size_t)btile * 2048;
    for (int i = tid; i < 2048; i += 128) {
      int r = i >> 7, c = i & 127;
      int4 vh = ghi[i];
      int4 vl = glo[i];
      *(int4*)(&sAhi[r * AROW + c * 8]) = vh;
      *(int4*)(&sAlo[r * AROW + c * 8]) = vl;
    }
  }
  __syncthreads();

  const int nt = wave * 64 + jt;  // gate n-tile (0..255 over 4096 gate columns)
  const int nloc = lane & 15;
  const int hh = lane >> 4;

  // accumulator init = x[t,b]*W_ih[g] + (b_ih+b_hh)[g]   (I == 1)
  float wv = Wih[nt * 16 + nloc];
  float bs = bsum[nt * 16 + nloc];
  v8f acc;
#pragma unroll
  for (int r = 0; r < 8; ++r) {
    int m = r + hh * 8;  // C/D layout: VGPR r -> M=r (lanes 0-15), M=r+8 (16-31)
    float xv = x[t * 64 + btile * 16 + m];
    acc[r] = xv * wv + bs;
  }

  // B base: flat int4 index (tn*32+kk)*32*2 + lane*2
  const int4* BH = (const int4*)Wphi + ((size_t)nt * 2048 + lane * 2);
  const int4* BL = (const int4*)Wplo + ((size_t)nt * 2048 + lane * 2);
  const unsigned short* Ab = &sAhi[(lane & 15) * AROW + 16 * hh];
  const unsigned short* Lb = &sAlo[(lane & 15) * AROW + 16 * hh];

#pragma unroll 4
  for (int kk = 0; kk < 32; ++kk) {
    Op32 bhi, blo, ahi, alo;
    bhi.q[0] = BH[(size_t)kk * 64];
    bhi.q[1] = BH[(size_t)kk * 64 + 1];
    blo.q[0] = BL[(size_t)kk * 64];
    blo.q[1] = BL[(size_t)kk * 64 + 1];
    ahi.q[0] = *(const int4*)(Ab + kk * 32);        // K-run [16*hh, +8)
    ahi.q[1] = *(const int4*)(Ab + kk * 32 + 8);    // K-run [16*hh+8, +8)
    alo.q[0] = *(const int4*)(Lb + kk * 32);
    alo.q[1] = *(const int4*)(Lb + kk * 32 + 8);
    acc = __builtin_amdgcn_wmma_f32_16x16x32_bf16(false, ahi.b, false, bhi.b,
                                                  (short)0, acc, false, false);
    acc = __builtin_amdgcn_wmma_f32_16x16x32_bf16(false, ahi.b, false, blo.b,
                                                  (short)0, acc, false, false);
    acc = __builtin_amdgcn_wmma_f32_16x16x32_bf16(false, alo.b, false, bhi.b,
                                                  (short)0, acc, false, false);
  }

  // ---- spill the 4 gate tiles to LDS for the cross-gate cell update ----
#pragma unroll
  for (int r = 0; r < 8; ++r) sG[wave * 256 + r * 32 + lane] = acc[r];
  __syncthreads();

  // ---- fused LSTM cell update: 2 (m,n) pairs per thread ----
#pragma unroll
  for (int q = 0; q < 2; ++q) {
    int p = tid * 2 + q;
    int m = p >> 4, n = p & 15;
    int idx = (m & 7) * 32 + n + ((m & 8) ? 16 : 0);
    float gi = sG[0 * 256 + idx];
    float gf = sG[1 * 256 + idx];
    float gg = sG[2 * 256 + idx];
    float go = sG[3 * 256 + idx];
    int off = (btile * 16 + m) * 1024 + jt * 16 + n;
    float co = cbuf[off];
    float iv = sigm(gi), fv = sigm(gf), ov = sigm(go);
    float gv = tanh_fast(gg);
    float cn = fv * co + iv * gv;
    float hn = ov * tanh_fast(cn);
    cbuf[off] = cn;
    unsigned short h_hi = f2bf(hn);
    hhiN[off] = h_hi;
    hloN[off] = f2bf(hn - bf2f(h_hi));
  }
}

// final head for out[511]
__global__ __launch_bounds__(128) void lstm_head(const unsigned short* __restrict__ hh,
                                                 const unsigned short* __restrict__ hl,
                                                 const float* __restrict__ Wout,
                                                 const float* __restrict__ bout,
                                                 float* __restrict__ out, int srow) {
  __shared__ float shh[128];
  int tid = threadIdx.x;
  int bl = tid >> 3, sub = tid & 7;
  int b = blockIdx.x * 16 + bl;
  const unsigned short* ph = hh + b * 1024;
  const unsigned short* pl = hl + b * 1024;
  float s = 0.f;
  for (int j = sub; j < 1024; j += 8) s += (bf2f(ph[j]) + bf2f(pl[j])) * Wout[j];
  shh[tid] = s;
  __syncthreads();
  if (sub == 0) {
    float tot = 0.f;
#pragma unroll
    for (int q = 0; q < 8; ++q) tot += shh[bl * 8 + q];
    out[srow * 64 + b] = tot + bout[0];
  }
}

extern "C" void kernel_launch(void* const* d_in, const int* in_sizes, int n_in,
                              void* d_out, int out_size, void* d_ws, size_t ws_size,
                              hipStream_t stream) {
  (void)in_sizes; (void)n_in; (void)out_size; (void)ws_size;
  const float* x    = (const float*)d_in[0];  // [512,64,1]
  const float* Wih  = (const float*)d_in[1];  // [4096,1]
  const float* Whh  = (const float*)d_in[2];  // [4096,1024]
  const float* bih  = (const float*)d_in[3];  // [4096]
  const float* bhh  = (const float*)d_in[4];  // [4096]
  const float* Wout = (const float*)d_in[5];  // [1,1024]
  const float* bout = (const float*)d_in[6];  // [1]
  const float* h0   = (const float*)d_in[7];  // [64,1024]
  const float* c0   = (const float*)d_in[8];  // [64,1024]
  float* out = (float*)d_out;                 // [512,64,1]

  char* ws = (char*)d_ws;
  unsigned short* Wphi = (unsigned short*)ws;               // 8 MB
  unsigned short* Wplo = (unsigned short*)(ws + 8388608);   // 8 MB
  float* bsum = (float*)(ws + 16777216);                    // 16 KB
  float* cb   = (float*)(ws + 16793600);                    // 256 KB
  unsigned short* hhiA = (unsigned short*)(ws + 17055744);  // 128 KB
  unsigned short* hloA = (unsigned short*)(ws + 17186816);  // 128 KB
  unsigned short* hhiB = (unsigned short*)(ws + 17317888);  // 128 KB
  unsigned short* hloB = (unsigned short*)(ws + 17448960);  // 128 KB

  lstm_pack<<<1024, 256, 0, stream>>>(Whh, Wphi, Wplo);
  lstm_init<<<528, 256, 0, stream>>>(bih, bhh, h0, c0, bsum, hhiA, hloA, cb);

  unsigned short *hhp = hhiA, *hlp = hloA, *hhn = hhiB, *hln = hloB;
  for (int t = 0; t < 512; ++t) {
    lstm_step<<<260, 128, 0, stream>>>(x, Wphi, Wplo, bsum, Wih, cb,
                                       hhp, hlp, hhn, hln, Wout, bout, out, t);
    unsigned short* tmp;
    tmp = hhp; hhp = hhn; hhn = tmp;
    tmp = hlp; hlp = hln; hln = tmp;
  }
  lstm_head<<<4, 128, 0, stream>>>(hhp, hlp, Wout, bout, out, 511);
}
